// WindowMamba3D_77953656422891
// MI455X (gfx1250) — matrix operand order repaired and checked by reading, not compiled, not run on hardware
//
#include <hip/hip_runtime.h>
#include <hip/hip_bf16.h>
#include <cstdint>
#include <cstddef>

#define D_MODEL 192
#define D_STATE 32
#define D_CONV  4
#define D_INNER 384              // 2*D_MODEL
#define DT_RANK 12
#define BATCH   256
#define SEQ     392
#define MROWS   (BATCH * SEQ)    // 100352
#define LN_EPS  1e-5f
#define NPAD    80               // x_proj output cols padded 76 -> 80
#define CONV_TL 8                // l-positions per conv thread (SEQ % 8 == 0)

typedef __bf16 bf16;
typedef __attribute__((ext_vector_type(16))) __bf16 v16bf;
typedef __attribute__((ext_vector_type(8)))  __bf16 v8bf;
typedef __attribute__((ext_vector_type(8)))  float  v8f;
typedef __attribute__((ext_vector_type(4)))  unsigned int v4u;
typedef __attribute__((ext_vector_type(8)))  int v8i;
typedef __attribute__((ext_vector_type(4)))  int v4i;

union FragBF { v16bf v; v8bf h[2]; };

__device__ __forceinline__ bf16 to_bf16(float x) { return (bf16)x; }

// ---------------------------------------------------------------------------
// LayerNorm (fp32 in) -> bf16 out. One wave32 per row of 192.
// ---------------------------------------------------------------------------
__global__ void ln_to_bf16(const float* __restrict__ x, const float* __restrict__ g,
                           const float* __restrict__ b, bf16* __restrict__ xn) {
  const int wave = threadIdx.x >> 5;
  const int lane = threadIdx.x & 31;
  const int row  = blockIdx.x * 8 + wave;
  const float* xr = x + (size_t)row * D_MODEL;
  float v[6];
  float s = 0.f, sq = 0.f;
#pragma unroll
  for (int i = 0; i < 6; ++i) {
    v[i] = xr[lane + i * 32];
    s += v[i]; sq += v[i] * v[i];
  }
#pragma unroll
  for (int m = 16; m >= 1; m >>= 1) {
    s  += __shfl_xor(s,  m, 32);
    sq += __shfl_xor(sq, m, 32);
  }
  const float mean = s * (1.f / D_MODEL);
  const float var  = sq * (1.f / D_MODEL) - mean * mean;
  const float rstd = rsqrtf(var + LN_EPS);
  bf16* outr = xn + (size_t)row * D_MODEL;
#pragma unroll
  for (int i = 0; i < 6; ++i) {
    const int e = lane + i * 32;
    outr[e] = to_bf16((v[i] - mean) * rstd * g[e] + b[e]);
  }
}

// ---------------------------------------------------------------------------
// bf16 WMMA GEMM:  C[M x N](f32) = A[M x K](bf16,row) * W[N x K](bf16,row)^T
// Block = 256 threads (8 waves). Wave tile = (MT*16) x (NT*16).
// The W tile for this block (NT*16 rows x K) is DMA'd into LDS once by the
// Tensor Data Mover (iterate mode: one K-row per iteration, LDS rows padded
// by 8 elements so b128 LDS reads are bank-conflict free), then all 8 waves
// feed their B fragments from ds_load_b128 while the A stream pipelines from
// global with next-K-step register prefetch.
// Fragment layouts per ISA 7.12.2:
//   A (16-bit 16x32): lanes 0-15 K[k0..7]+K[k0+16..23]; lanes 16-31 +8.
//   B: lane = column; lanes 0-15 hold K[k0..7]+K[k0+16..23], lanes 16-31 +8 (same as A).
//   D: VGPR r -> row r (lanes 0-15) / 8+r (lanes 16-31), lane%16 = column.
// ---------------------------------------------------------------------------
template <int MT, int NT, int K>
__global__ void gemm_bf16_wmma(const bf16* __restrict__ A, const bf16* __restrict__ W,
                               const float* __restrict__ bias, float* __restrict__ C,
                               int ldc) {
  constexpr int ROWS = NT * 16;
  constexpr int LDSK = K + 8;                    // padded LDS row stride (elems)
  __shared__ bf16 wtile[ROWS * LDSK];

  const int wave = threadIdx.x >> 5;
  const int lane = threadIdx.x & 31;
  const int lm   = lane & 15;
  const int hi   = lane >> 4;        // half-wave: 0 or 1
  const int m0   = (blockIdx.x * 8 + wave) * (MT * 16);
  const int n0   = blockIdx.y * ROWS;

  // ---- TDM: async-copy W tile (ROWS x K bf16) -> LDS, one wave issues ----
  if (wave == 0) {
    const unsigned long long ga = (unsigned long long)(const void*)(W + (size_t)n0 * K);
    const unsigned int       la = (unsigned int)(uintptr_t)(void*)wtile;
    v4u g0 = { 1u,                                    // count=1 (valid user D#)
               la,                                    // lds_addr (bytes)
               (unsigned int)ga,                      // global_addr[31:0]
               (unsigned int)((ga >> 32) & 0x1FFFFFFu) | (2u << 30) };  // type=2
    v8i g1 = { (int)((1u << 16) | (1u << 19)),        // data_size=2B, iterate_enable
               (int)(((unsigned)K & 0xFFFFu) << 16),  // tensor_dim0[15:0]
               (int)((((unsigned)K >> 16) & 0xFFFFu) | (((unsigned)ROWS & 0xFFFFu) << 16)),
               (int)(((unsigned)K & 0xFFFFu) << 16),  // tile_dim0 = K
               1,                                     // tile_dim1 = 1 row / iter
               (int)K,                                // tensor_dim0_stride[31:0]
               0, 0 };
    v4i g2 = { 0,                                     // tensor_dim2
               (int)LDSK,                             // lds_addr_increment (elems)
               (int)K,                                // global_addr_increment (elems)
               (int)(((unsigned)(ROWS - 1) & 0xFFFFu) << 16) };  // iterate_count
    v4i g3 = { 0, 0, 0, 0 };
    v8i g4 = { 0, 0, 0, 0, 0, 0, 0, 0 };              // unused trailing group
    __builtin_amdgcn_tensor_load_to_lds(g0, g1, g2, g3, g4, 0);
    __builtin_amdgcn_s_wait_tensorcnt(0);
  }
  __syncthreads();

  v8f acc[MT][NT] = {};
  const bf16* abase = A + (size_t)(m0 + lm) * K + hi * 8;
  const bf16* lbase = wtile + (size_t)lm * LDSK + hi * 8;

  FragBF af[MT];
#pragma unroll
  for (int t = 0; t < MT; ++t) {
    const bf16* ap = abase + (size_t)t * 16 * K;
    af[t].h[0] = *(const v8bf*)(ap);
    af[t].h[1] = *(const v8bf*)(ap + 16);
  }

#pragma unroll
  for (int k0 = 0; k0 < K; k0 += 32) {
    FragBF an[MT];
    if (k0 + 32 < K) {              // prefetch next K-step A frags from global
#pragma unroll
      for (int t = 0; t < MT; ++t) {
        const bf16* ap = abase + (size_t)t * 16 * K + k0 + 32;
        an[t].h[0] = *(const v8bf*)(ap);
        an[t].h[1] = *(const v8bf*)(ap + 16);
      }
    }
    FragBF bf[NT];
#pragma unroll
    for (int j = 0; j < NT; ++j) {   // B frags from LDS (bank-conflict free)
      const bf16* wp = lbase + (size_t)j * 16 * LDSK + k0;
      bf[j].h[0] = *(const v8bf*)(wp);
      bf[j].h[1] = *(const v8bf*)(wp + 16);
    }
#pragma unroll
    for (int t = 0; t < MT; ++t)
#pragma unroll
      for (int j = 0; j < NT; ++j)
        acc[t][j] = __builtin_amdgcn_wmma_f32_16x16x32_bf16(
            false, af[t].v, false, bf[j].v, (short)0, acc[t][j], false, false);
    if (k0 + 32 < K) {
#pragma unroll
      for (int t = 0; t < MT; ++t) af[t] = an[t];
    }
  }

#pragma unroll
  for (int j = 0; j < NT; ++j) {
    const int   n  = n0 + j * 16 + lm;
    const float bv = bias ? bias[n] : 0.f;
#pragma unroll
    for (int t = 0; t < MT; ++t) {
#pragma unroll
      for (int r = 0; r < 8; ++r) {
        const int m = m0 + t * 16 + hi * 8 + r;
        C[(size_t)m * ldc + n] = acc[t][j][r] + bv;
      }
    }
  }
}

// ---------------------------------------------------------------------------
// Depthwise causal conv (4 taps) + bias + SiLU, register sliding window.
// Thread handles CONV_TL consecutive l for one (b,d): 11 loads -> 8 outputs.
// ---------------------------------------------------------------------------
__global__ void conv_silu(const float* __restrict__ xz, const float* __restrict__ cw,
                          const float* __restrict__ cb, float* __restrict__ u,
                          bf16* __restrict__ ubf) {
  const int d   = threadIdx.x;
  const int blk = blockIdx.x;
  const int b   = blk / (SEQ / CONV_TL);
  const int l0  = (blk % (SEQ / CONV_TL)) * CONV_TL;

  float w[D_CONV];
#pragma unroll
  for (int j = 0; j < D_CONV; ++j) w[j] = cw[d * D_CONV + j];
  const float bias = cb[d];

  float win[CONV_TL + D_CONV - 1];
#pragma unroll
  for (int j = 0; j < CONV_TL + D_CONV - 1; ++j) {
    const int l = l0 + j - (D_CONV - 1);
    win[j] = (l >= 0) ? xz[((size_t)(b * SEQ + l)) * (2 * D_INNER) + d] : 0.f;
  }
#pragma unroll
  for (int i = 0; i < CONV_TL; ++i) {
    float acc = bias;
#pragma unroll
    for (int j = 0; j < D_CONV; ++j) acc += w[j] * win[i + j];
    const float s = acc / (1.f + __expf(-acc));
    const size_t o = ((size_t)(b * SEQ + l0 + i)) * D_INNER + d;
    u[o]   = s;
    ubf[o] = to_bf16(s);
  }
}

// ---------------------------------------------------------------------------
// delta = softplus(dt @ dt_proj_w^T + dt_proj_b).  One block per row.
// ---------------------------------------------------------------------------
__global__ void delta_softplus(const float* __restrict__ dbl, const float* __restrict__ dtw,
                               const float* __restrict__ dtb, float* __restrict__ delta) {
  const int row = blockIdx.x;
  const int d   = threadIdx.x;
  __shared__ float dts[DT_RANK];
  if (d < DT_RANK) dts[d] = dbl[(size_t)row * NPAD + d];
  __syncthreads();
  float s = dtb[d];
#pragma unroll
  for (int r = 0; r < DT_RANK; ++r) s += dts[r] * dtw[d * DT_RANK + r];
  const float sp = (s > 20.f) ? s : __logf(1.f + __expf(s));
  delta[(size_t)row * D_INNER + d] = sp;
}

// ---------------------------------------------------------------------------
// Selective scan. One block per batch, one thread per inner channel d.
// h[32] and A-row live in VGPRs; B/C broadcast via double-buffered LDS
// (one barrier per step). Fused epilogue: +u*D_skip, *silu(z), -> bf16.
// ---------------------------------------------------------------------------
__global__ void ssm_scan(const float* __restrict__ dbl, const float* __restrict__ delta,
                         const float* __restrict__ u, const float* __restrict__ xz,
                         const float* __restrict__ A_log, const float* __restrict__ Dskip,
                         bf16* __restrict__ ybf) {
  const int b = blockIdx.x;
  const int d = threadIdx.x;
  __shared__ float sh[2][64];   // [phase][0:32)=B_t, [32:64)=C_t

  float Arow[D_STATE];
#pragma unroll
  for (int n = 0; n < D_STATE; ++n) Arow[n] = -__expf(A_log[d * D_STATE + n]);
  const float dsk = Dskip[d];

  float h[D_STATE];
#pragma unroll
  for (int n = 0; n < D_STATE; ++n) h[n] = 0.f;

  for (int l = 0; l < SEQ; ++l) {
    const size_t row = (size_t)b * SEQ + l;
    const int    p   = l & 1;
    if (d < 64) sh[p][d] = dbl[row * NPAD + DT_RANK + d];
    __syncthreads();
    const float dt = delta[row * D_INNER + d];
    const float uv = u[row * D_INNER + d];
    const float du = dt * uv;
    float y = 0.f;
#pragma unroll
    for (int n = 0; n < D_STATE; ++n) {
      const float dA = __expf(dt * Arow[n]);
      h[n] = dA * h[n] + du * sh[p][n];
      y += h[n] * sh[p][32 + n];
    }
    const float zv = xz[row * (2 * D_INNER) + D_INNER + d];
    const float yv = (y + uv * dsk) * (zv / (1.f + __expf(-zv)));
    ybf[row * D_INNER + d] = to_bf16(yv);
  }
}

// ---------------------------------------------------------------------------
// Weight prep kernels.
// ---------------------------------------------------------------------------
__global__ void cvt_f32_bf16(const float* __restrict__ in, bf16* __restrict__ out, int n) {
  const int i = blockIdx.x * blockDim.x + threadIdx.x;
  if (i < n) out[i] = to_bf16(in[i]);
}

__global__ void cvt_xproj_pad(const float* __restrict__ in, bf16* __restrict__ out) {
  const int i = blockIdx.x * blockDim.x + threadIdx.x;
  if (i >= NPAD * D_INNER) return;
  const int n = i / D_INNER, k = i % D_INNER;
  out[i] = (n < (DT_RANK + 2 * D_STATE)) ? to_bf16(in[n * D_INNER + k]) : to_bf16(0.f);
}

// W_fused[c][d] = sum_e proj_w[c][e] * out_proj_w[e][d]   (192 x 384)
__global__ void fuse_out_weights(const float* __restrict__ pw, const float* __restrict__ ow,
                                 bf16* __restrict__ wf) {
  const int idx = blockIdx.x * blockDim.x + threadIdx.x;
  if (idx >= D_MODEL * D_INNER) return;
  const int c = idx / D_INNER, d = idx % D_INNER;
  float acc = 0.f;
  for (int e = 0; e < D_MODEL; ++e) acc += pw[c * D_MODEL + e] * ow[e * D_INNER + d];
  wf[idx] = to_bf16(acc);
}

// ---------------------------------------------------------------------------
extern "C" void kernel_launch(void* const* d_in, const int* in_sizes, int n_in,
                              void* d_out, int out_size, void* d_ws, size_t ws_size,
                              hipStream_t stream) {
  const float* x         = (const float*)d_in[0];
  const float* ln_g      = (const float*)d_in[1];
  const float* ln_b      = (const float*)d_in[2];
  const float* in_proj_w = (const float*)d_in[3];
  const float* conv_w    = (const float*)d_in[4];
  const float* conv_b    = (const float*)d_in[5];
  const float* x_proj_w  = (const float*)d_in[6];
  const float* dt_proj_w = (const float*)d_in[7];
  const float* dt_proj_b = (const float*)d_in[8];
  const float* A_log     = (const float*)d_in[9];
  const float* D_skip    = (const float*)d_in[10];
  const float* out_proj_w= (const float*)d_in[11];
  const float* proj_w    = (const float*)d_in[12];
  const float* proj_b    = (const float*)d_in[13];
  float* out = (float*)d_out;

  // ---- workspace layout (256B aligned) ----
  char* base = (char*)d_ws;
  size_t off = 0;
  auto take = [&](size_t bytes) -> char* {
    char* p = base + off;
    off = (off + bytes + 255) & ~(size_t)255;
    return p;
  };
  bf16*  w_in   = (bf16*) take((size_t)(2 * D_INNER) * D_MODEL * 2);   // 768x192
  bf16*  w_xp   = (bf16*) take((size_t)NPAD * D_INNER * 2);            // 80x384 (padded)
  bf16*  w_fuse = (bf16*) take((size_t)D_MODEL * D_INNER * 2);         // 192x384
  bf16*  xn     = (bf16*) take((size_t)MROWS * D_MODEL * 2);
  float* xz     = (float*)take((size_t)MROWS * (2 * D_INNER) * 4);
  float* u_f    = (float*)take((size_t)MROWS * D_INNER * 4);
  bf16*  u_bf   = (bf16*) take((size_t)MROWS * D_INNER * 2);
  float* dbl    = (float*)take((size_t)MROWS * NPAD * 4);
  float* delta  = (float*)take((size_t)MROWS * D_INNER * 4);
  bf16*  y_bf   = u_bf;  // alias: u_bf16 dead after x_proj GEMM

  // ---- weight prep ----
  {
    const int n1 = 2 * D_INNER * D_MODEL;  // 147456
    cvt_f32_bf16<<<(n1 + 255) / 256, 256, 0, stream>>>(in_proj_w, w_in, n1);
    const int n2 = NPAD * D_INNER;         // 30720
    cvt_xproj_pad<<<(n2 + 255) / 256, 256, 0, stream>>>(x_proj_w, w_xp);
    const int n3 = D_MODEL * D_INNER;      // 73728
    fuse_out_weights<<<(n3 + 255) / 256, 256, 0, stream>>>(proj_w, out_proj_w, w_fuse);
  }

  // ---- 1) LayerNorm -> bf16 ----
  ln_to_bf16<<<MROWS / 8, 256, 0, stream>>>(x, ln_g, ln_b, xn);

  // ---- 2) in_proj GEMM: (M x 192) x (192 -> 768) ----
  gemm_bf16_wmma<4, 2, D_MODEL><<<dim3(MROWS / 512, (2 * D_INNER) / 32), 256, 0, stream>>>(
      xn, w_in, nullptr, xz, 2 * D_INNER);

  // ---- 3) depthwise causal conv + SiLU ----
  conv_silu<<<BATCH * (SEQ / CONV_TL), D_INNER, 0, stream>>>(xz, conv_w, conv_b, u_f, u_bf);

  // ---- 4) x_proj GEMM: (M x 384) x (384 -> 80 padded) ----
  gemm_bf16_wmma<4, 1, D_INNER><<<dim3(MROWS / 512, NPAD / 16), 256, 0, stream>>>(
      u_bf, w_xp, nullptr, dbl, NPAD);

  // ---- 5) delta = softplus(dt @ dt_proj_w^T + b) ----
  delta_softplus<<<MROWS, D_INNER, 0, stream>>>(dbl, dt_proj_w, dt_proj_b, delta);

  // ---- 6) selective scan + gated epilogue -> y (bf16) ----
  ssm_scan<<<BATCH, D_INNER, 0, stream>>>(dbl, delta, u_f, xz, A_log, D_skip, y_bf);

  // ---- 7) fused output GEMM: (M x 384) x (384 -> 192) + proj_b ----
  gemm_bf16_wmma<4, 2, D_INNER><<<dim3(MROWS / 512, D_MODEL / 32), 256, 0, stream>>>(
      y_bf, w_fuse, proj_b, out, D_MODEL);

  (void)in_sizes; (void)n_in; (void)out_size; (void)ws_size;
}
